// CGT_23459111371193
// MI455X (gfx1250) — compile-verified, matrix operand order checked by reading, not run on hardware
//
#include <hip/hip_runtime.h>

// ---------------------------------------------------------------------------
// CGT forward for MI455X (gfx1250, wave32). Big GEMMs -> v_wmma_f32_16x16x32_bf16
// with async global->LDS staging (ASYNCcnt) for A tiles and a transposed-B LDS
// layout so fragments load as ds_load_b128. Scatter/segment ops use f32 atomics.
// ---------------------------------------------------------------------------

#define NN 10000   // nodes
#define EE 80000   // edges
#define BB 64      // graphs

typedef __attribute__((ext_vector_type(16))) __bf16        bf16x16;
typedef __attribute__((ext_vector_type(8)))  float         floatx8;
typedef __attribute__((ext_vector_type(8)))  unsigned int  uintx8;

__device__ __forceinline__ unsigned short f2bf(float f) {
  unsigned int u = __builtin_bit_cast(unsigned int, f);
  u += 0x7FFFu + ((u >> 16) & 1u);           // round-to-nearest-even
  return (unsigned short)(u >> 16);
}
__device__ __forceinline__ float bf2f(unsigned short h) {
  unsigned int u = ((unsigned int)h) << 16;
  return __builtin_bit_cast(float, u);
}
// monotone float atomic max via integer ordering trick (init must be -inf)
__device__ __forceinline__ void atomicMaxF(float* addr, float val) {
  if (val >= 0.0f) atomicMax((int*)addr, __float_as_int(val));
  else             atomicMin((unsigned int*)addr, __float_as_uint(val));
}
__device__ __forceinline__ void wait_asynccnt0() {
#if __has_builtin(__builtin_amdgcn_s_wait_asynccnt)
  __builtin_amdgcn_s_wait_asynccnt(0);
#else
  asm volatile("s_wait_asynccnt 0x0" ::: "memory");
#endif
}

// ---------------------------------------------------------------------------
// Utility kernels
// ---------------------------------------------------------------------------
__global__ void fill_kernel(float* p, float v, int n) {
  int i = blockIdx.x * blockDim.x + threadIdx.x;
  if (i < n) p[i] = v;
}

__global__ void tobf16_kernel(const float* __restrict__ in,
                              unsigned short* __restrict__ out, int n) {
  int i = blockIdx.x * blockDim.x + threadIdx.x;
  if (i < n) out[i] = f2bf(in[i]);
}

__global__ void embed_kernel(const int* __restrict__ ids,
                             const float* __restrict__ emb,
                             float* __restrict__ x,
                             unsigned short* __restrict__ xb) {
  int i = blockIdx.x * blockDim.x + threadIdx.x;
  if (i >= NN * 256) return;
  int n = i >> 8, c = i & 255;
  float v = emb[(size_t)ids[n] * 256 + c];
  x[i] = v;
  xb[i] = f2bf(v);
}

// Small dense layer (f32 in/out), one thread per output element.
__global__ void dense_kernel(const float* __restrict__ A,
                             const float* __restrict__ W,
                             const float* __restrict__ bias,
                             float* __restrict__ out,
                             int M, int N, int K, int lrelu) {
  int i = blockIdx.x * blockDim.x + threadIdx.x;
  if (i >= M * N) return;
  int m = i / N, n = i % N;
  float s = bias ? bias[n] : 0.0f;
  const float* a = A + (size_t)m * K;
  for (int k = 0; k < K; ++k) s += a[k] * W[(size_t)k * N + n];
  if (lrelu) s = s > 0.0f ? s : 0.01f * s;
  out[i] = s;
}

// Dense layer writing bf16 (edge MLP layer 1, K=14 too small for WMMA).
__global__ void dense_bf16out_kernel(const float* __restrict__ A,
                                     const float* __restrict__ W,
                                     const float* __restrict__ bias,
                                     unsigned short* __restrict__ out,
                                     int M, int N, int K, int lrelu) {
  int i = blockIdx.x * blockDim.x + threadIdx.x;
  if (i >= M * N) return;
  int m = i / N, n = i % N;
  float s = bias ? bias[n] : 0.0f;
  const float* a = A + (size_t)m * K;
  for (int k = 0; k < K; ++k) s += a[k] * W[(size_t)k * N + n];
  if (lrelu) s = s > 0.0f ? s : 0.01f * s;
  out[i] = f2bf(s);
}

// ---------------------------------------------------------------------------
// WMMA GEMM: C[M,N] = A[M,K](bf16) * B[K,N](bf16) (+bias) (+lrelu)
// 256 threads = 8 waves; block tile 32x64 (2 M-tiles x 4 N-tiles per block).
// A tile staged via global_load_async_to_lds_b64 (ASYNCcnt path); B tile staged
// transposed (BsT[n][k]) so both fragments read as two ds_load_b128 per wave.
// flags: bit0 = leaky relu, bit1 = write f32 to Cf, bit2 = write bf16 to Cb
// Requires: K % 32 == 0, N % 64 == 0 (M may be ragged; rows clamped/guarded).
// ---------------------------------------------------------------------------
__global__ __launch_bounds__(256)
void gemm_bf16_wmma(const unsigned short* __restrict__ A,
                    const unsigned short* __restrict__ B,
                    const float* __restrict__ bias,
                    float* __restrict__ Cf,
                    unsigned short* __restrict__ Cb,
                    int M, int N, int K, int flags) {
  __shared__ unsigned short As[32][40];    // 80B row stride: b128 reads aligned
  __shared__ unsigned short BsT[64][40];   // transposed B tile, same alignment

  const int tid  = threadIdx.x;
  const int wave = tid >> 5;
  const int lane = tid & 31;
  const int wm   = wave >> 2;              // 0..1 (M tile within block)
  const int wn   = wave & 3;               // 0..3 (N tile within block)
  const int mblk = blockIdx.y * 32;
  const int nblk = blockIdx.x * 64;
  const int hl   = lane >> 4;              // K-half select per ISA A/B layout
  const int lr   = lane & 15;

  // Precompute per-thread staging coordinates.
  const int ar = (tid * 4) >> 5, ac = (tid * 4) & 31;   // A: 4 u16 (8B) / thread
  const int br = (tid * 8) >> 6, bc = (tid * 8) & 63;   // B: 8 u16 (16B) / thread
  int arow = mblk + ar; if (arow > M - 1) arow = M - 1; // clamp ragged M
  const unsigned aLds = (unsigned)(size_t)&As[ar][ac];  // LDS aperture is in the
                                                        // high dword -> low 32
                                                        // bits == LDS offset

  floatx8 acc = {0.f, 0.f, 0.f, 0.f, 0.f, 0.f, 0.f, 0.f};

#pragma unroll 2
  for (int k0 = 0; k0 < K; k0 += 32) {
    // ---- A tile: per-lane async DMA global -> LDS (8 bytes each) ----
    {
      const unsigned short* ga = A + (size_t)arow * K + (k0 + ac);
      asm volatile("global_load_async_to_lds_b64 %0, %1, off"
                   :: "v"(aLds), "v"(ga) : "memory");
    }
    // ---- B tile: coalesced 16B global load, transposed scatter into LDS ----
    {
      const unsigned short* gb = B + (size_t)(k0 + br) * N + (nblk + bc);
      uint4 d = *(const uint4*)gb;
      unsigned w[4] = {d.x, d.y, d.z, d.w};
#pragma unroll
      for (int j = 0; j < 8; ++j)
        BsT[bc + j][br] = (unsigned short)(w[j >> 1] >> ((j & 1) * 16));
    }
    if (k0 + 32 < K) {                     // pull next tiles toward the caches
      __builtin_prefetch(A + (size_t)arow * K + (k0 + 32 + ac), 0, 0);
      __builtin_prefetch(B + (size_t)(k0 + 32 + br) * N + (nblk + bc), 0, 0);
    }
    wait_asynccnt0();                      // async LDS writes visible
    __syncthreads();

    // ---- fragments: ISA 16-bit 16x32 layout == two contiguous b128 reads ----
    const int am = wm * 16 + lr;
    const int bn = wn * 16 + lr;
    uint4 a0 = *(const uint4*)&As[am][hl * 8];
    uint4 a1 = *(const uint4*)&As[am][16 + hl * 8];
    uint4 b0 = *(const uint4*)&BsT[bn][hl * 8];
    uint4 b1 = *(const uint4*)&BsT[bn][16 + hl * 8];
    uintx8 au = {a0.x, a0.y, a0.z, a0.w, a1.x, a1.y, a1.z, a1.w};
    uintx8 bu = {b0.x, b0.y, b0.z, b0.w, b1.x, b1.y, b1.z, b1.w};
    bf16x16 af = __builtin_bit_cast(bf16x16, au);
    bf16x16 bf = __builtin_bit_cast(bf16x16, bu);
    acc = __builtin_amdgcn_wmma_f32_16x16x32_bf16(false, af, false, bf,
                                                  (short)0, acc, false, false);
    __syncthreads();
  }

  // ---- epilogue per C/D layout: lane -> N, vgpr -> M (+8 for lanes 16-31) --
  const int n = nblk + wn * 16 + lr;
  float bv = bias ? bias[n] : 0.0f;
#pragma unroll
  for (int r = 0; r < 8; ++r) {
    int m = mblk + wm * 16 + hl * 8 + r;
    if (m >= M) continue;
    float vv = acc[r] + bv;
    if (flags & 1) vv = vv > 0.0f ? vv : 0.01f * vv;
    if (flags & 2) Cf[(size_t)m * N + n] = vv;
    if (flags & 4) Cb[(size_t)m * N + n] = f2bf(vv);
  }
}

// ---------------------------------------------------------------------------
// Attention kernels (wave per (edge, head))
// ---------------------------------------------------------------------------
__global__ void edge_alpha_kernel(const int* __restrict__ ei,
                                  const float* __restrict__ q,
                                  const float* __restrict__ k,
                                  const unsigned short* __restrict__ ep,
                                  float* __restrict__ alpha,
                                  float* __restrict__ mbuf) {
  int gt = blockIdx.x * blockDim.x + threadIdx.x;
  int gw = gt >> 5, lane = gt & 31;
  if (gw >= EE * 4) return;
  int e = gw >> 2, h = gw & 3;
  int src = ei[e], tgt = ei[EE + e];
  const float* qp = q + (size_t)tgt * 1024 + h * 256;
  const float* kp = k + (size_t)src * 1024 + h * 256;
  const unsigned short* epp = ep + (size_t)e * 1024 + h * 256;
  float s = 0.0f;
#pragma unroll
  for (int j = 0; j < 8; ++j) {
    int c = lane + j * 32;
    s += qp[c] * (kp[c] + bf2f(epp[c]));
  }
  for (int off = 16; off; off >>= 1) s += __shfl_xor(s, off, 32);
  if (lane == 0) {
    float a = s * 0.0625f;                 // 1/sqrt(C), C=256
    alpha[(size_t)e * 4 + h] = a;
    atomicMaxF(&mbuf[(size_t)tgt * 4 + h], a);
  }
}

__global__ void edge_exp_kernel(const int* __restrict__ ei,
                                float* __restrict__ alpha,
                                const float* __restrict__ mbuf,
                                float* __restrict__ denom) {
  int i = blockIdx.x * blockDim.x + threadIdx.x;
  if (i >= EE * 4) return;
  int e = i >> 2, h = i & 3;
  int tgt = ei[EE + e];
  float ex = expf(alpha[i] - mbuf[(size_t)tgt * 4 + h]);
  alpha[i] = ex;
  atomicAdd(&denom[(size_t)tgt * 4 + h], ex);
}

__global__ void edge_msg_kernel(const int* __restrict__ ei,
                                const float* __restrict__ v,
                                const unsigned short* __restrict__ ep,
                                const float* __restrict__ alpha,
                                const float* __restrict__ denom,
                                float* __restrict__ agg) {
  int gt = blockIdx.x * blockDim.x + threadIdx.x;
  int gw = gt >> 5, lane = gt & 31;
  if (gw >= EE * 4) return;
  int e = gw >> 2, h = gw & 3;
  int src = ei[e], tgt = ei[EE + e];
  float a = alpha[(size_t)e * 4 + h] / (denom[(size_t)tgt * 4 + h] + 1e-16f);
  const float* vp = v + (size_t)src * 1024 + h * 256;
  const unsigned short* epp = ep + (size_t)e * 1024 + h * 256;
  float* ap = agg + (size_t)tgt * 1024 + h * 256;
#pragma unroll
  for (int j = 0; j < 8; ++j) {
    int c = lane + j * 32;
    atomicAdd(&ap[c], (vp[c] + bf2f(epp[c])) * a);
  }
}

__global__ void finish_kernel(const float* __restrict__ agg,
                              const float* __restrict__ xskip,
                              float* __restrict__ x,
                              unsigned short* __restrict__ xb) {
  int i = blockIdx.x * blockDim.x + threadIdx.x;
  if (i >= NN * 256) return;
  int n = i >> 8, c = i & 255;
  const float* a = agg + (size_t)n * 1024 + c;
  float s = 0.25f * (a[0] + a[256] + a[512] + a[768]) + xskip[i];
  x[i] = s;
  xb[i] = f2bf(s);
}

// ---------------------------------------------------------------------------
// Pooling + head
// ---------------------------------------------------------------------------
__global__ void pool_kernel(const float* __restrict__ x,
                            const int* __restrict__ batch,
                            float* __restrict__ sump,
                            float* __restrict__ maxp,
                            float* __restrict__ cnt) {
  int i = blockIdx.x * blockDim.x + threadIdx.x;
  if (i >= NN * 256) return;
  int n = i >> 8, c = i & 255;
  int b = batch[n];
  float v = x[i];
  atomicAdd(&sump[(size_t)b * 256 + c], v);
  atomicMaxF(&maxp[(size_t)b * 256 + c], v);
  if (c == 0) atomicAdd(&cnt[b], 1.0f);
}

__global__ void feat_kernel(const float* __restrict__ sump,
                            const float* __restrict__ maxp,
                            const float* __restrict__ cnt,
                            const float* __restrict__ en,
                            float* __restrict__ feat) {
  int i = blockIdx.x * blockDim.x + threadIdx.x;
  if (i >= BB * 896) return;
  int b = i / 896, k = i % 896;
  float v;
  if (k < 256)      v = sump[b * 256 + k] / cnt[b];
  else if (k < 512) v = maxp[b * 256 + (k - 256)];
  else if (k < 768) v = sump[b * 256 + (k - 512)];
  else              v = en[b * 128 + (k - 768)];
  feat[i] = v;
}

// ---------------------------------------------------------------------------
// Host orchestration
// ---------------------------------------------------------------------------
static inline int cdiv(int a, int b) { return (a + b - 1) / b; }

static inline void launch_gemm(const unsigned short* A, const unsigned short* B,
                               const float* bias, float* Cf, unsigned short* Cb,
                               int M, int N, int K, int flags, hipStream_t s) {
  dim3 grid(N / 64, cdiv(M, 32));
  gemm_bf16_wmma<<<grid, dim3(256), 0, s>>>(A, B, bias, Cf, Cb, M, N, K, flags);
}

extern "C" void kernel_launch(void* const* d_in, const int* in_sizes, int n_in,
                              void* d_out, int out_size, void* d_ws, size_t ws_size,
                              hipStream_t stream) {
  (void)in_sizes; (void)n_in; (void)out_size; (void)ws_size;

  const int*   x_ids    = (const int*)  d_in[0];
  const int*   eidx     = (const int*)  d_in[1];   // [2,E] : row0=src, row1=tgt
  const int*   batch    = (const int*)  d_in[2];
  const float* edge_attr= (const float*)d_in[3];
  const float* energies = (const float*)d_in[4];
  const float* node_emb = (const float*)d_in[5];
  const float* edge_w1  = (const float*)d_in[6];
  const float* edge_b1  = (const float*)d_in[7];
  const float* edge_w2  = (const float*)d_in[8];
  const float* edge_b2  = (const float*)d_in[9];
  const float* wq       = (const float*)d_in[10];
  const float* wk       = (const float*)d_in[11];
  const float* wv       = (const float*)d_in[12];
  const float* we       = (const float*)d_in[13];
  const float* wskip    = (const float*)d_in[14];
  const float* fce_w1   = (const float*)d_in[15];
  const float* fce_b1   = (const float*)d_in[16];
  const float* fce_w2   = (const float*)d_in[17];
  const float* fce_b2   = (const float*)d_in[18];
  const float* fc_w1    = (const float*)d_in[19];
  const float* fc_b1    = (const float*)d_in[20];
  const float* fc_w2    = (const float*)d_in[21];
  const float* fc_b2    = (const float*)d_in[22];
  float*       out      = (float*)d_out;

  // ---- workspace carve-out (256B aligned) ----
  size_t off = 0;
  char* base = (char*)d_ws;
  auto alloc = [&](size_t elems, size_t esz) -> char* {
    char* p = base + off;
    off = (off + elems * esz + 255) & ~(size_t)255;
    return p;
  };
  float*          x     = (float*)         alloc((size_t)NN * 256, 4);
  unsigned short* xb    = (unsigned short*)alloc((size_t)NN * 256, 2);
  float*          xskip = (float*)         alloc((size_t)NN * 256, 4);
  float*          q     = (float*)         alloc((size_t)NN * 1024, 4);
  float*          kbuf  = (float*)         alloc((size_t)NN * 1024, 4);
  float*          vbuf  = (float*)         alloc((size_t)NN * 1024, 4);
  unsigned short* eah   = (unsigned short*)alloc((size_t)EE * 128, 2);
  unsigned short* eab   = (unsigned short*)alloc((size_t)EE * 256, 2);
  unsigned short* eproj = (unsigned short*)alloc((size_t)EE * 1024, 2);
  float*          alpha = (float*)         alloc((size_t)EE * 4, 4);
  float*          mbuf  = (float*)         alloc((size_t)NN * 4, 4);
  float*          denom = (float*)         alloc((size_t)NN * 4, 4);
  float*          agg   = (float*)         alloc((size_t)NN * 1024, 4);
  unsigned short* wbq   = (unsigned short*)alloc(256 * 1024, 2);
  unsigned short* wbk   = (unsigned short*)alloc(256 * 1024, 2);
  unsigned short* wbv   = (unsigned short*)alloc(256 * 1024, 2);
  unsigned short* wbe   = (unsigned short*)alloc(256 * 1024, 2);
  unsigned short* wbs   = (unsigned short*)alloc(256 * 256, 2);
  unsigned short* wb2   = (unsigned short*)alloc(128 * 256, 2);
  float*          sump  = (float*)         alloc(BB * 256, 4);
  float*          maxp  = (float*)         alloc(BB * 256, 4);
  float*          cnt   = (float*)         alloc(BB, 4);
  float*          enh   = (float*)         alloc(BB * 256, 4);
  float*          en    = (float*)         alloc(BB * 128, 4);
  float*          feat  = (float*)         alloc(BB * 896, 4);
  float*          h1    = (float*)         alloc(BB * 1024, 4);

  const int T = 256;
  const float NEG_INF = -__builtin_huge_valf();

  // ---- embedding + edge MLP ----
  embed_kernel<<<cdiv(NN * 256, T), T, 0, stream>>>(x_ids, node_emb, x, xb);
  tobf16_kernel<<<cdiv(128 * 256, T), T, 0, stream>>>(edge_w2, wb2, 128 * 256);
  dense_bf16out_kernel<<<cdiv(EE * 128, T), T, 0, stream>>>(
      edge_attr, edge_w1, edge_b1, eah, EE, 128, 14, 1);
  launch_gemm(eah, wb2, edge_b2, nullptr, eab, EE, 256, 128, /*bf16 out*/4, stream);

  // ---- 3 transformer layers ----
  for (int l = 0; l < 3; ++l) {
    tobf16_kernel<<<cdiv(256 * 1024, T), T, 0, stream>>>(wq + (size_t)l * 256 * 1024, wbq, 256 * 1024);
    tobf16_kernel<<<cdiv(256 * 1024, T), T, 0, stream>>>(wk + (size_t)l * 256 * 1024, wbk, 256 * 1024);
    tobf16_kernel<<<cdiv(256 * 1024, T), T, 0, stream>>>(wv + (size_t)l * 256 * 1024, wbv, 256 * 1024);
    tobf16_kernel<<<cdiv(256 * 1024, T), T, 0, stream>>>(we + (size_t)l * 256 * 1024, wbe, 256 * 1024);
    tobf16_kernel<<<cdiv(256 * 256, T), T, 0, stream>>>(wskip + (size_t)l * 256 * 256, wbs, 256 * 256);

    launch_gemm(xb, wbq, nullptr, q,    nullptr, NN, 1024, 256, 2, stream);
    launch_gemm(xb, wbk, nullptr, kbuf, nullptr, NN, 1024, 256, 2, stream);
    launch_gemm(xb, wbv, nullptr, vbuf, nullptr, NN, 1024, 256, 2, stream);
    launch_gemm(eab, wbe, nullptr, nullptr, eproj, EE, 1024, 256, 4, stream);
    launch_gemm(xb, wbs, nullptr, xskip, nullptr, NN, 256, 256, 2, stream);

    fill_kernel<<<cdiv(NN * 4, T), T, 0, stream>>>(mbuf, NEG_INF, NN * 4);
    fill_kernel<<<cdiv(NN * 4, T), T, 0, stream>>>(denom, 0.0f, NN * 4);
    fill_kernel<<<cdiv(NN * 1024, T), T, 0, stream>>>(agg, 0.0f, NN * 1024);

    edge_alpha_kernel<<<cdiv(EE * 4 * 32, T), T, 0, stream>>>(eidx, q, kbuf, eproj, alpha, mbuf);
    edge_exp_kernel<<<cdiv(EE * 4, T), T, 0, stream>>>(eidx, alpha, mbuf, denom);
    edge_msg_kernel<<<cdiv(EE * 4 * 32, T), T, 0, stream>>>(eidx, vbuf, eproj, alpha, denom, agg);
    finish_kernel<<<cdiv(NN * 256, T), T, 0, stream>>>(agg, xskip, x, xb);
  }

  // ---- pooling ----
  fill_kernel<<<cdiv(BB * 256, T), T, 0, stream>>>(sump, 0.0f, BB * 256);
  fill_kernel<<<cdiv(BB * 256, T), T, 0, stream>>>(maxp, NEG_INF, BB * 256);
  fill_kernel<<<1, T, 0, stream>>>(cnt, 0.0f, BB);
  pool_kernel<<<cdiv(NN * 256, T), T, 0, stream>>>(x, batch, sump, maxp, cnt);

  // ---- head MLPs (B=64: VALU is fine) ----
  dense_kernel<<<cdiv(BB * 256, T), T, 0, stream>>>(energies, fce_w1, fce_b1, enh, BB, 256, 201, 1);
  dense_kernel<<<cdiv(BB * 128, T), T, 0, stream>>>(enh, fce_w2, fce_b2, en, BB, 128, 256, 0);
  feat_kernel<<<cdiv(BB * 896, T), T, 0, stream>>>(sump, maxp, cnt, en, feat);
  dense_kernel<<<cdiv(BB * 1024, T), T, 0, stream>>>(feat, fc_w1, fc_b1, h1, BB, 1024, 896, 1);
  dense_kernel<<<cdiv(BB * 804, T), T, 0, stream>>>(h1, fc_w2, fc_b2, out, BB, 804, 1024, 0);
}
